// NonLocalBlock_89799176225336
// MI455X (gfx1250) — matrix operand order repaired and checked by reading, not compiled, not run on hardware
//
#include <hip/hip_runtime.h>

typedef __attribute__((ext_vector_type(16))) _Float16 v16h;
typedef __attribute__((ext_vector_type(8)))  _Float16 v8h;
typedef __attribute__((ext_vector_type(8)))  float    v8f;

#define BB   8
#define CC   256
#define NN   4096
#define NS   1024
#define CIC  128
#define EPSV 1e-5f

#define WMMA_F16(a, b, c) \
  __builtin_amdgcn_wmma_f32_16x16x32_f16(false, (a), false, (b), (short)0, (c), false, false)

// ---- Async global->LDS copy (gfx1250 ASYNCcnt path), with sync fallback -----
#if defined(__has_builtin)
#  if __has_builtin(__builtin_amdgcn_global_load_async_to_lds_b128) && \
      __has_builtin(__builtin_amdgcn_s_wait_asynccnt)
#    define HAVE_ASYNC 1
#  endif
#endif
#ifndef HAVE_ASYNC
#  define HAVE_ASYNC 0
#endif

// Exact parameter types per hipcc diagnostic: pointer to GNU-vector int4 in
// AS1 (printed '__device__') for the global source, AS3 for the LDS dest.
typedef int v4i_gnu __attribute__((vector_size(16)));
typedef __attribute__((address_space(1))) v4i_gnu* gv4p;
typedef __attribute__((address_space(3))) v4i_gnu* sv4p;

__device__ __forceinline__ void copy16(_Float16* l, const _Float16* g) {
#if HAVE_ASYNC
  __builtin_amdgcn_global_load_async_to_lds_b128((gv4p)(void*)g, (sv4p)(void*)l, 0, 0);
#else
  *(v8h*)l = *(const v8h*)g;
#endif
}
__device__ __forceinline__ void copy_fence() {
#if HAVE_ASYNC
  __builtin_amdgcn_s_wait_asynccnt(0);
#endif
}

// ---- Fragment loaders -------------------------------------------------------
// A-matrix (16x32 f16, MxK): lane L holds row (L&15); K runs [sel*8, +8) and
// [sel*8+16, +8), sel = L>>4 (ISA 7.12.2 A-table).
__device__ __forceinline__ v16h frag_a_h(const _Float16* base, int ld, int row0, int k0) {
  const int lane = threadIdx.x & 31;
  const int sel  = lane >> 4;
  const _Float16* p = base + (size_t)(row0 + (lane & 15)) * ld + k0 + sel * 8;
  union { v16h v; struct { v8h lo, hi; } h; } u;
  u.h.lo = *(const v8h*)(p);
  u.h.hi = *(const v8h*)(p + 16);
  return u.v;
}

// B-matrix (32x16 f16, KxN): lane L holds column (L&15); K runs [sel*8, +8)
// and [sel*8+16, +8) (matches A-table). Source is B^T row-major ([col][k]).
__device__ __forceinline__ v16h frag_b_h(const _Float16* base, int ld, int col0, int k0) {
  const int lane = threadIdx.x & 31;
  const int sel  = lane >> 4;
  const _Float16* p = base + (size_t)(col0 + (lane & 15)) * ld + k0 + sel * 8;
  union { v16h v; struct { v8h lo, hi; } h; } u;
  u.h.lo = *(const v8h*)(p);
  u.h.hi = *(const v8h*)(p + 16);
  return u.v;
}

// ---- Prep 0a: weights f32 -> f16 (all four are 128*256 = 32768 elements) ----
__global__ __launch_bounds__(256) void wconv_kernel(
    const float* __restrict__ tw, const float* __restrict__ pw,
    const float* __restrict__ gw, const float* __restrict__ ww,
    _Float16* __restrict__ twh, _Float16* __restrict__ pwh,
    _Float16* __restrict__ gwh, _Float16* __restrict__ wwh)
{
  const int i = blockIdx.x * 256 + threadIdx.x;
  twh[i] = (_Float16)tw[i];
  pwh[i] = (_Float16)pw[i];
  gwh[i] = (_Float16)gw[i];
  wwh[i] = (_Float16)ww[i];
}

// ---- Prep 0b: x [B][C][N] f32 -> xT [B][N][C] f16 (tiled transpose) ---------
__global__ __launch_bounds__(256) void xpose_kernel(
    const float* __restrict__ x, _Float16* __restrict__ xT)
{
  __shared__ float tile[32][33];
  const int b = blockIdx.z;
  const int n0 = blockIdx.x * 32, c0 = blockIdx.y * 32;
  const int t0 = threadIdx.x & 31, t1 = threadIdx.x >> 5;   // 8 rows / pass
  const float* xb = x + (size_t)b * CC * NN;
#pragma unroll
  for (int i = 0; i < 32; i += 8)
    tile[t1 + i][t0] = xb[(size_t)(c0 + t1 + i) * NN + n0 + t0];
  __syncthreads();
  _Float16* xo = xT + (size_t)b * NN * CC;
#pragma unroll
  for (int i = 0; i < 32; i += 8)
    xo[(size_t)(n0 + t1 + i) * CC + c0 + t0] = (_Float16)tile[t0][t1 + i];
}

// ---- Kernel 1: 1x1 conv projections (theta / phi / g) -----------------------
// grid (32 n-tiles, 3 proj, 8 batch); 8 waves, each a 16(ci) x 128(n) strip.
__global__ __launch_bounds__(256) void proj_kernel(
    const _Float16* __restrict__ xT,
    const _Float16* __restrict__ twh, const float* __restrict__ tb,
    const _Float16* __restrict__ pwh, const float* __restrict__ pb,
    const _Float16* __restrict__ gwh, const float* __restrict__ gb,
    _Float16* __restrict__ thetaH, float* __restrict__ phiF, float* __restrict__ gF)
{
  __shared__ _Float16 Bs[128][136];  // x tile, [n][c] (already transposed)
  const int b = blockIdx.z, proj = blockIdx.y;
  const int n0 = blockIdx.x * 128;
  const int tid = threadIdx.x;
  const int wave = tid >> 5, lane = tid & 31, sel = lane >> 4, lc = lane & 15;
  const int m0 = wave * 16;

  const _Float16* wmat = (proj == 0) ? twh : ((proj == 1) ? pwh : gwh);
  const float*    bias = (proj == 0) ? tb : ((proj == 1) ? pb : gb);
  const _Float16* xb = xT + (size_t)b * NN * CC;

  v8f acc[8];
#pragma unroll
  for (int i = 0; i < 8; ++i)
#pragma unroll
    for (int j = 0; j < 8; ++j) acc[i][j] = 0.0f;

  for (int kk = 0; kk < CC; kk += 128) {
    __syncthreads();
#pragma unroll
    for (int i = 0; i < 8; ++i) {           // 128 rows x 256B = 2048 x 16B
      const int e = tid + i * 256;
      const int n = e >> 4, q = e & 15;
      copy16(&Bs[n][q * 8], xb + (size_t)(n0 + n) * CC + kk + q * 8);
    }
    copy_fence();
    __syncthreads();
#pragma unroll
    for (int kt = 0; kt < 4; ++kt) {
      const v16h a = frag_a_h(wmat, CC, m0, kk + kt * 32);
#pragma unroll
      for (int nt = 0; nt < 8; ++nt) {
        const v16h bf = frag_b_h(&Bs[0][0], 136, nt * 16, kt * 32);
        acc[nt] = WMMA_F16(a, bf, acc[nt]);
      }
    }
  }

  float bv[8];
#pragma unroll
  for (int j = 0; j < 8; ++j) bv[j] = bias[m0 + j + 8 * sel];

  if (proj == 0) {
    _Float16* out = thetaH + (size_t)b * NN * CIC;
#pragma unroll
    for (int nt = 0; nt < 8; ++nt) {
      const int n = n0 + nt * 16 + lc;
#pragma unroll
      for (int j = 0; j < 8; ++j)
        out[(size_t)n * CIC + m0 + j + 8 * sel] = (_Float16)(acc[nt][j] + bv[j]);
    }
  } else {
    float* out = ((proj == 1) ? phiF : gF) + (size_t)b * CIC * NN;
#pragma unroll
    for (int nt = 0; nt < 8; ++nt) {
#pragma unroll
      for (int j = 0; j < 8; ++j)
        out[(size_t)(m0 + j + 8 * sel) * NN + n0 + nt * 16 + lc] = acc[nt][j] + bv[j];
    }
  }
}

// ---- Kernel 2: 2x2 maxpool, f32 -> f16 with attention-friendly layouts ------
__global__ __launch_bounds__(256) void pool_kernel(
    const float* __restrict__ phiF, const float* __restrict__ gF,
    _Float16* __restrict__ phiT, _Float16* __restrict__ gP)
{
  const int idx = blockIdx.x * 256 + threadIdx.x;       // 8*128*1024 total
  const int ns = idx & 1023;
  const int ci = (idx >> 10) & 127;
  const int b  = idx >> 17;
  const int h2 = ns >> 5, w2 = ns & 31;
  const size_t base = (size_t)(b * CIC + ci) * NN + (size_t)(h2 * 2) * 64 + w2 * 2;
  const float pm = fmaxf(fmaxf(phiF[base], phiF[base + 1]),
                         fmaxf(phiF[base + 64], phiF[base + 65]));
  const float gm = fmaxf(fmaxf(gF[base], gF[base + 1]),
                         fmaxf(gF[base + 64], gF[base + 65]));
  phiT[((size_t)b * NS + ns) * CIC + ci] = (_Float16)pm;     // [b][ns][ci]
  gP[(size_t)(b * CIC + ci) * NS + ns]   = (_Float16)gm;     // [b][ci][ns]
}

// ---- Kernel 3: fused attention (flash-style online softmax) -----------------
__global__ __launch_bounds__(256) void attn_kernel(
    const _Float16* __restrict__ thetaH, const _Float16* __restrict__ phiT,
    const _Float16* __restrict__ gP, _Float16* __restrict__ yH)
{
  __shared__ _Float16 PhiS[32][136];   // [m][ci]
  __shared__ _Float16 GS[128][40];     // [ci][m]
  __shared__ _Float16 Ps[8][16][40];   // per-wave P tile [row][m]
  const int b = blockIdx.y;
  const int n0 = blockIdx.x * 128;
  const int tid = threadIdx.x;
  const int wave = tid >> 5, lane = tid & 31, sel = lane >> 4, lc = lane & 15;

  const _Float16* th = thetaH + (size_t)b * NN * CIC;
  const _Float16* ph = phiT + (size_t)b * NS * CIC;
  const _Float16* gp = gP + (size_t)b * CIC * NS;

  v16h af[4];
#pragma unroll
  for (int kt = 0; kt < 4; ++kt) af[kt] = frag_a_h(th, CIC, n0 + wave * 16, kt * 32);

  v8f yacc[8];
  float rm[8], rl[8];
#pragma unroll
  for (int j = 0; j < 8; ++j) {
    rm[j] = -1e30f; rl[j] = 0.0f;
#pragma unroll
    for (int nt = 0; nt < 8; ++nt) yacc[nt][j] = 0.0f;
  }

  for (int ms = 0; ms < NS; ms += 32) {
    __syncthreads();
#pragma unroll
    for (int i = 0; i < 2; ++i) {          // PhiS: 32 rows x 256B
      const int e = tid + i * 256;
      const int mm = e >> 4, q = e & 15;
      copy16(&PhiS[mm][q * 8], ph + (size_t)(ms + mm) * CIC + q * 8);
    }
#pragma unroll
    for (int i = 0; i < 2; ++i) {          // GS: 128 rows x 64B
      const int e = tid + i * 256;
      const int ci = e >> 2, q = e & 3;
      copy16(&GS[ci][q * 8], gp + (size_t)ci * NS + ms + q * 8);
    }
    copy_fence();
    __syncthreads();

    v8f s0, s1;
#pragma unroll
    for (int j = 0; j < 8; ++j) { s0[j] = 0.0f; s1[j] = 0.0f; }
#pragma unroll
    for (int kt = 0; kt < 4; ++kt) {
      s0 = WMMA_F16(af[kt], frag_b_h(&PhiS[0][0], 136, 0,  kt * 32), s0);
      s1 = WMMA_F16(af[kt], frag_b_h(&PhiS[0][0], 136, 16, kt * 32), s1);
    }

    _Float16* pbuf = &Ps[wave][0][0];
#pragma unroll
    for (int j = 0; j < 8; ++j) {
      float v = fmaxf(s0[j], s1[j]);
#pragma unroll
      for (int off = 1; off < 16; off <<= 1) v = fmaxf(v, __shfl_xor(v, off, 32));
      const float nm = fmaxf(rm[j], v);
      const float scale = __expf(rm[j] - nm);
      rm[j] = nm;
      const float p0 = __expf(s0[j] - nm);
      const float p1 = __expf(s1[j] - nm);
      float rs = p0 + p1;
#pragma unroll
      for (int off = 1; off < 16; off <<= 1) rs += __shfl_xor(rs, off, 32);
      rl[j] = rl[j] * scale + rs;
#pragma unroll
      for (int nt = 0; nt < 8; ++nt) yacc[nt][j] *= scale;
      const int row = j + 8 * sel;
      pbuf[row * 40 + lc]      = (_Float16)p0;
      pbuf[row * 40 + 16 + lc] = (_Float16)p1;
    }

    const v16h ap = frag_a_h(pbuf, 40, 0, 0);
#pragma unroll
    for (int nt = 0; nt < 8; ++nt)
      yacc[nt] = WMMA_F16(ap, frag_b_h(&GS[0][0], 40, nt * 16, 0), yacc[nt]);
  }

  _Float16* yo = yH + (size_t)b * NN * CIC;
#pragma unroll
  for (int j = 0; j < 8; ++j) {
    const int n = n0 + wave * 16 + j + 8 * sel;
    const float inv = 1.0f / rl[j];
#pragma unroll
    for (int nt = 0; nt < 8; ++nt)
      yo[(size_t)n * CIC + nt * 16 + lc] = (_Float16)(yacc[nt][j] * inv);
  }
}

// ---- Kernel 4: wy = W*y + b, plus deterministic per-block BN partials -------
__global__ __launch_bounds__(256) void wgemm_kernel(
    const _Float16* __restrict__ yH, const _Float16* __restrict__ wwh,
    const float* __restrict__ wb, float* __restrict__ wy,
    float* __restrict__ pSum, float* __restrict__ pSq)
{
  const int b = blockIdx.z, mh = blockIdx.y;
  const int n0 = blockIdx.x * 128;
  const int tid = threadIdx.x;
  const int wave = tid >> 5, lane = tid & 31, sel = lane >> 4, lc = lane & 15;
  const int m0 = mh * 128 + wave * 16;
  const int slot = b * 32 + blockIdx.x;          // 256 slots, disjoint m per mh
  const _Float16* yb = yH + (size_t)b * NN * CIC;

  v8f acc[8];
#pragma unroll
  for (int i = 0; i < 8; ++i)
#pragma unroll
    for (int j = 0; j < 8; ++j) acc[i][j] = 0.0f;

#pragma unroll
  for (int kt = 0; kt < 4; ++kt) {
    const v16h a = frag_a_h(wwh, CIC, m0, kt * 32);
#pragma unroll
    for (int nt = 0; nt < 8; ++nt)
      acc[nt] = WMMA_F16(a, frag_b_h(yb, CIC, n0 + nt * 16, kt * 32), acc[nt]);
  }

  float* wyb = wy + (size_t)b * CC * NN;
#pragma unroll
  for (int j = 0; j < 8; ++j) {
    const int m = m0 + j + 8 * sel;
    const float bias = wb[m];
    float s1 = 0.0f, s2 = 0.0f;
#pragma unroll
    for (int nt = 0; nt < 8; ++nt) {
      const float v = acc[nt][j] + bias;
      wyb[(size_t)m * NN + n0 + nt * 16 + lc] = v;
      s1 += v; s2 += v * v;
    }
#pragma unroll
    for (int off = 1; off < 16; off <<= 1) {
      s1 += __shfl_xor(s1, off, 32);
      s2 += __shfl_xor(s2, off, 32);
    }
    if (lc == 0) {
      pSum[(size_t)slot * CC + m] = s1;
      pSq[(size_t)slot * CC + m]  = s2;
    }
  }
}

// ---- Kernel 5: reduce BN partials -> stats[sum | sumsq] ---------------------
__global__ __launch_bounds__(256) void stats_reduce(
    const float* __restrict__ pSum, const float* __restrict__ pSq,
    float* __restrict__ stats)
{
  __shared__ float sm1[256], sm2[256];
  const int c = blockIdx.x, t = threadIdx.x;
  sm1[t] = pSum[(size_t)t * CC + c];
  sm2[t] = pSq[(size_t)t * CC + c];
  __syncthreads();
  for (int s = 128; s > 0; s >>= 1) {
    if (t < s) { sm1[t] += sm1[t + s]; sm2[t] += sm2[t + s]; }
    __syncthreads();
  }
  if (t == 0) { stats[c] = sm1[0]; stats[CC + c] = sm2[0]; }
}

// ---- Kernel 6: BatchNorm (training stats) + residual ------------------------
__global__ __launch_bounds__(256) void bn_kernel(
    const float* __restrict__ wy, const float* __restrict__ stats,
    const float* __restrict__ gamma, const float* __restrict__ beta,
    const float* __restrict__ x, float* __restrict__ out)
{
  const size_t idx = (size_t)blockIdx.x * 256 + threadIdx.x;
  const int c = (int)((idx >> 12) & 255);
  const float inv = 1.0f / (float)(BB * NN);
  const float mean = stats[c] * inv;
  const float var  = stats[CC + c] * inv - mean * mean;
  const float v = (wy[idx] - mean) * rsqrtf(var + EPSV);
  out[idx] = v * gamma[c] + beta[c] + x[idx];
}

// ---- Host launcher ----------------------------------------------------------
extern "C" void kernel_launch(void* const* d_in, const int* in_sizes, int n_in,
                              void* d_out, int out_size, void* d_ws, size_t ws_size,
                              hipStream_t stream)
{
  (void)in_sizes; (void)n_in; (void)out_size; (void)ws_size;
  const float* x       = (const float*)d_in[0];
  const float* g_w     = (const float*)d_in[1];
  const float* g_b     = (const float*)d_in[2];
  const float* theta_w = (const float*)d_in[3];
  const float* theta_b = (const float*)d_in[4];
  const float* phi_w   = (const float*)d_in[5];
  const float* phi_b   = (const float*)d_in[6];
  const float* w_w     = (const float*)d_in[7];
  const float* w_b     = (const float*)d_in[8];
  const float* bn_g    = (const float*)d_in[9];
  const float* bn_b    = (const float*)d_in[10];
  float* out = (float*)d_out;

  char* ws = (char*)d_ws;
  size_t off = 0;
  auto take = [&](size_t bytes) -> void* {
    void* p = ws + off;
    off += (bytes + 255) & ~(size_t)255;
    return p;
  };
  _Float16* xT     = (_Float16*)take((size_t)BB * NN * CC * 2);    // 16 MB
  _Float16* thetaH = (_Float16*)take((size_t)BB * NN * CIC * 2);   // 8 MB
  float*    phiF   = (float*)take((size_t)BB * CIC * NN * 4);      // 16 MB
  float*    gF     = (float*)take((size_t)BB * CIC * NN * 4);      // 16 MB
  _Float16* phiT   = (_Float16*)take((size_t)BB * NS * CIC * 2);   // 2 MB
  _Float16* gP     = (_Float16*)take((size_t)BB * CIC * NS * 2);   // 2 MB
  _Float16* yH     = (_Float16*)take((size_t)BB * NN * CIC * 2);   // 8 MB
  float*    wy     = (float*)take((size_t)BB * CC * NN * 4);       // 32 MB
  float*    pSum   = (float*)take((size_t)256 * CC * 4);
  float*    pSq    = (float*)take((size_t)256 * CC * 4);
  float*    stats  = (float*)take((size_t)2 * CC * 4);
  _Float16* twh    = (_Float16*)take((size_t)CIC * CC * 2);
  _Float16* pwh    = (_Float16*)take((size_t)CIC * CC * 2);
  _Float16* gwh    = (_Float16*)take((size_t)CIC * CC * 2);
  _Float16* wwh    = (_Float16*)take((size_t)CC * CIC * 2);

  wconv_kernel<<<dim3(128), 256, 0, stream>>>(theta_w, phi_w, g_w, w_w,
                                              twh, pwh, gwh, wwh);
  xpose_kernel<<<dim3(128, 8, 8), 256, 0, stream>>>(x, xT);
  proj_kernel<<<dim3(32, 3, 8), 256, 0, stream>>>(
      xT, twh, theta_b, pwh, phi_b, gwh, g_b, thetaH, phiF, gF);
  pool_kernel<<<dim3(4096), 256, 0, stream>>>(phiF, gF, phiT, gP);
  attn_kernel<<<dim3(32, 8), 256, 0, stream>>>(thetaH, phiT, gP, yH);
  wgemm_kernel<<<dim3(32, 2, 8), 256, 0, stream>>>(yH, wwh, w_b, wy, pSum, pSq);
  stats_reduce<<<dim3(256), 256, 0, stream>>>(pSum, pSq, stats);
  bn_kernel<<<dim3(32768), 256, 0, stream>>>(wy, stats, bn_g, bn_b, x, out);
}
